// DecoderRNN_69672959476055
// MI455X (gfx1250) — compile-verified, matrix operand order checked by reading, not run on hardware
//
#include <hip/hip_runtime.h>
#include <math.h>

typedef __attribute__((ext_vector_type(16))) __bf16 v16bf;
typedef __attribute__((ext_vector_type(8)))  float  v8f;

// ---------------------------------------------------------------------------
// Generic NT GEMM: C[M=128,N] = A[128,K] * B[N,K]^T (+bias), f32 in/out,
// bf16 WMMA (v_wmma_f32_16x16x32_bf16) with f32 accumulation.
// Block = 256 threads (8 waves). Wave w owns M-rows [16w,16w+16);
// block owns N-columns [64*blockIdx.x, +64) as 4 16-wide subtiles.
// Software pipelining:
//   - next k-step's 64x64 f32 W tile is global-loaded into registers while
//     the current tile's 8 WMMAs execute (hides HBM latency of the dominant
//     205 MB W_out stream), converted via v_cvt_pk_bf16_f32 and stored to
//     LDS after the tail barrier;
//   - B fragments are double-buffered so ds_load latency of fragment tt+1
//     overlaps WMMA tt (partial s_wait_dscnt instead of full serialization).
// ---------------------------------------------------------------------------
__global__ void __launch_bounds__(256)
gemm_nt_bf16(const float* __restrict__ A, const float* __restrict__ Bm,
             const float* __restrict__ bias, float* __restrict__ C,
             int N, int K) {
  __shared__ __bf16 Bt[64][72];                  // 64 k-cols + pad
  const int tid  = threadIdx.x;
  const int lane = tid & 31;
  const int wave = tid >> 5;                     // 0..7 -> m-tile
  const int n0   = blockIdx.x * 64;
  const int l16  = lane & 15;
  const int base = (lane < 16) ? 0 : 8;          // ISA A/B k-phase split
  const int mrow = (wave << 4) + l16;            // A row for this lane

  v8f acc[4];
#pragma unroll
  for (int t = 0; t < 4; ++t)
#pragma unroll
    for (int rr = 0; rr < 8; ++rr) acc[t][rr] = 0.f;

  // staging assignment: thread -> (row r, quad q); 4 float4 per k-step
  const int r = tid >> 2;                        // 0..63
  const int q = tid & 3;                         // 0..3
  int nld = n0 + r; if (nld >= N) nld = N - 1;   // clamp; masked at store
  const float* brow = Bm + (size_t)nld * K;
  const float* arow = A + (size_t)mrow * K;

  // preload k-step 0 tile into registers
  float4 y[4];
#pragma unroll
  for (int j = 0; j < 4; ++j)
    y[j] = *reinterpret_cast<const float4*>(brow + q * 4 + j * 16);

  for (int k0 = 0; k0 < K; k0 += 64) {
    // ---- store staged tile to LDS (f32 -> bf16 hardware cvt)
#pragma unroll
    for (int j = 0; j < 4; ++j) {
      int cc = q * 4 + j * 16;
      Bt[r][cc + 0] = (__bf16)y[j].x;
      Bt[r][cc + 1] = (__bf16)y[j].y;
      Bt[r][cc + 2] = (__bf16)y[j].z;
      Bt[r][cc + 3] = (__bf16)y[j].w;
    }
    __syncthreads();

    // ---- issue next tile's global loads now; they complete during WMMAs
    float4 yn[4];
    if (k0 + 64 < K) {
#pragma unroll
      for (int j = 0; j < 4; ++j)
        yn[j] = *reinterpret_cast<const float4*>(brow + k0 + 64 + q * 4 + j * 16);
    }

#pragma unroll
    for (int kh = 0; kh < 2; ++kh) {
      const int kc = kh * 32 + base;
      // ---- A fragment: lane = row mrow, k in {base..+7, base+16..+23}
      v16bf a;
      {
        const float* ap = arow + k0 + kh * 32 + base;
        float4 x0 = *reinterpret_cast<const float4*>(ap);
        float4 x1 = *reinterpret_cast<const float4*>(ap + 4);
        float4 x2 = *reinterpret_cast<const float4*>(ap + 16);
        float4 x3 = *reinterpret_cast<const float4*>(ap + 20);
        a[0]  = (__bf16)x0.x; a[1]  = (__bf16)x0.y; a[2]  = (__bf16)x0.z; a[3]  = (__bf16)x0.w;
        a[4]  = (__bf16)x1.x; a[5]  = (__bf16)x1.y; a[6]  = (__bf16)x1.z; a[7]  = (__bf16)x1.w;
        a[8]  = (__bf16)x2.x; a[9]  = (__bf16)x2.y; a[10] = (__bf16)x2.z; a[11] = (__bf16)x2.w;
        a[12] = (__bf16)x3.x; a[13] = (__bf16)x3.y; a[14] = (__bf16)x3.z; a[15] = (__bf16)x3.w;
      }
      // ---- double-buffered B fragments: load tt+1 while WMMA tt runs
      auto ldfrag = [&](v16bf& f, int tt) {
        const int nr = (tt << 4) + l16;
#pragma unroll
        for (int e = 0; e < 8; ++e) f[e]     = Bt[nr][kc + e];
#pragma unroll
        for (int e = 0; e < 8; ++e) f[8 + e] = Bt[nr][kc + 16 + e];
      };
      v16bf buf[2];
      ldfrag(buf[0], 0);
#pragma unroll
      for (int tt = 0; tt < 4; ++tt) {
        if (tt < 3) ldfrag(buf[(tt + 1) & 1], tt + 1);
        acc[tt] = __builtin_amdgcn_wmma_f32_16x16x32_bf16(
            false, a, false, buf[tt & 1], (short)0, acc[tt], false, false);
      }
    }
    __syncthreads();
#pragma unroll
    for (int j = 0; j < 4; ++j) y[j] = yn[j];
  }

  // ---- epilogue: D layout lane = N, VGPR r -> M = r (+8 for upper lanes)
#pragma unroll
  for (int t = 0; t < 4; ++t) {
    int n = n0 + (t << 4) + l16;
    if (n < N) {
      float bv = bias ? bias[n] : 0.f;
#pragma unroll
      for (int rr = 0; rr < 8; ++rr) {
        int m = (wave << 4) + base + rr;
        C[(size_t)m * N + n] = acc[t][rr] + bv;
      }
    }
  }
}

// ---------------------------------------------------------------------------
// GRU elementwise: r,z,n gates -> h_new (also into combined[:, :H])
// ---------------------------------------------------------------------------
__global__ void gru_elem(const float* __restrict__ gi, const float* __restrict__ gh,
                         const float* __restrict__ h, float* __restrict__ out_h,
                         float* __restrict__ comb, int B, int H) {
  int i = blockIdx.x * blockDim.x + threadIdx.x;
  if (i >= B * H) return;
  int b = i / H, j = i - b * H;
  const float* gib = gi + (size_t)b * 3 * H;
  const float* ghb = gh + (size_t)b * 3 * H;
  float rr = 1.f / (1.f + __expf(-(gib[j] + ghb[j])));
  float z = 1.f / (1.f + __expf(-(gib[H + j] + ghb[H + j])));
  float n = tanhf(gib[2 * H + j] + rr * ghb[2 * H + j]);
  float hv = h[(size_t)b * H + j];
  float hn = (1.f - z) * n + z * hv;
  out_h[(size_t)b * H + j] = hn;
  comb[(size_t)b * 2 * H + j] = hn;
}

// v = h_new @ W_bil[0]   (NN, tiny: 128x512x512)
__global__ void bil_v(const float* __restrict__ hn, const float* __restrict__ Wb,
                      float* __restrict__ v, int H) {
  int b = blockIdx.x;
  for (int h = threadIdx.x; h < H; h += blockDim.x) {
    float acc = 0.f;
    for (int k = 0; k < H; ++k) acc += hn[(size_t)b * H + k] * Wb[(size_t)k * H + h];
    v[(size_t)b * H + h] = acc;
  }
}

// ---------------------------------------------------------------------------
// Fused attention (one pass over encoder_states): energy + online softmax +
// context accumulation. Grid = B*4 (4 s-chunks per batch row for occupancy),
// block = 512 threads (16 waves). enc tile 16 x 512 f32 staged in LDS (32KB).
// Partials (ctx[H], m, Z) per chunk go to ws; energies to ws_e.
// ---------------------------------------------------------------------------
__global__ void __launch_bounds__(512)
attn_fused(const float* __restrict__ v, const float* __restrict__ enc,
           const float* __restrict__ b_bil, float* __restrict__ e_out,
           float* __restrict__ part, int S, int B, int H, int SC) {
  __shared__ float tile[16][512];
  __shared__ float esh[16];
  const int b = blockIdx.x >> 2, c = blockIdx.x & 3;
  const int t = threadIdx.x;
  const int lane = t & 31, wv = t >> 5;
  const int s0c = c * SC;
  const int s1c = min(S, s0c + SC);
  const float* vp = v + (size_t)b * H;
  const float bb = b_bil[0];
  float m = -INFINITY, Z = 0.f, ctx = 0.f;

  for (int s0 = s0c; s0 < s1c; s0 += 16) {
    const int ns = min(16, s1c - s0);
    for (int j = 0; j < ns; ++j)
      tile[j][t] = enc[((size_t)(s0 + j) * B + b) * H + t];
    __syncthreads();
    if (wv < ns) {                      // wave wv -> energy of row s0+wv
      float sum = 0.f;
      for (int k = lane; k < H; k += 32) sum += vp[k] * tile[wv][k];
#pragma unroll
      for (int off = 16; off; off >>= 1) sum += __shfl_xor(sum, off, 32);
      if (lane == 0) {
        float e = sum + bb;
        esh[wv] = e;
        e_out[(size_t)b * S + s0 + wv] = e;
      }
    }
    __syncthreads();
    // online softmax update (all threads redundantly track identical m,Z)
    float tm = m;
    for (int j = 0; j < ns; ++j) tm = fmaxf(tm, esh[j]);
    float sc = __expf(m - tm);          // first tile: exp(-inf)=0 zeroes ctx,Z
    ctx *= sc; Z *= sc; m = tm;
    for (int j = 0; j < ns; ++j) {
      float w = __expf(esh[j] - m);
      Z += w;
      ctx += w * tile[j][t];
    }
    __syncthreads();
  }
  // partial layout: [B][4][H+2]
  float* pp = part + ((size_t)b * 4 + c) * (H + 2);
  pp[t] = ctx;
  if (t == 0) { pp[H] = m; pp[H + 1] = Z; }
}

// merge 4 chunk partials -> context (combined[:,H:2H]) and attn output
__global__ void __launch_bounds__(512)
attn_combine(const float* __restrict__ part, const float* __restrict__ e,
             float* __restrict__ attn, float* __restrict__ comb, int S, int H) {
  const int b = blockIdx.x, t = threadIdx.x;
  float mc[4], Zc[4];
  float m = -INFINITY;
#pragma unroll
  for (int c = 0; c < 4; ++c) {
    const float* pp = part + ((size_t)b * 4 + c) * (H + 2);
    mc[c] = pp[H]; Zc[c] = pp[H + 1];
    m = fmaxf(m, mc[c]);
  }
  float Z = 0.f, ctx = 0.f;
#pragma unroll
  for (int c = 0; c < 4; ++c) {
    float w = __expf(mc[c] - m);        // empty chunk: exp(-inf)=0
    Z += Zc[c] * w;
    ctx += part[((size_t)b * 4 + c) * (H + 2) + t] * w;
  }
  comb[(size_t)b * 2 * H + H + t] = ctx / Z;
  for (int s = t; s < S; s += 512)
    attn[(size_t)b * S + s] = __expf(e[(size_t)b * S + s] - m) / Z;
}

// prob_ptr[b] = sigmoid(combined[b] . W_ptr + b_ptr)
__global__ void pptr_k(const float* __restrict__ comb, const float* __restrict__ Wp,
                       const float* __restrict__ bp, float* __restrict__ p, int K2) {
  __shared__ float red[256];
  int b = blockIdx.x, t = threadIdx.x;
  float acc = 0.f;
  for (int k = t; k < K2; k += 256) acc += comb[(size_t)b * K2 + k] * Wp[k];
  red[t] = acc; __syncthreads();
  for (int o = 128; o; o >>= 1) { if (t < o) red[t] += red[t + o]; __syncthreads(); }
  if (t == 0) p[b] = 1.f / (1.f + __expf(-(red[0] + bp[0])));
}

// online row max + sum(exp) over V for softmax of logits (stored in d_out)
__global__ void rowstats_k(const float* __restrict__ logits, float* __restrict__ st, int V) {
  __shared__ float rm[256], rs[256];
  int b = blockIdx.x, t = threadIdx.x;
  float m = -INFINITY, s = 0.f;
  for (int v = t; v < V; v += 256) {
    float x = logits[(size_t)b * V + v];
    if (x > m) { s = s * __expf(m - x) + 1.f; m = x; }
    else       { s += __expf(x - m); }
  }
  rm[t] = m; rs[t] = s; __syncthreads();
  for (int o = 128; o; o >>= 1) {
    if (t < o) {
      float m2 = rm[t + o], s2 = rs[t + o];
      float M = fmaxf(rm[t], m2);
      rs[t] = rs[t] * __expf(rm[t] - M) + s2 * __expf(m2 - M);
      rm[t] = M;
    }
    __syncthreads();
  }
  if (t == 0) { st[2 * b] = rm[0]; st[2 * b + 1] = rs[0]; }
}

// in-place: out = (1-p[b]) * softmax(logits)
__global__ void finalize_k(float* __restrict__ out, const float* __restrict__ st,
                           const float* __restrict__ p, int V, long total) {
  long i = (long)blockIdx.x * blockDim.x + threadIdx.x;
  if (i >= total) return;
  int b = (int)(i / V);
  float x = out[i];
  out[i] = (1.f - p[b]) * __expf(x - st[2 * b]) / st[2 * b + 1];
}

// scatter-add pointer mass at encoder word ids
__global__ void scatter_k(float* __restrict__ out, const int* __restrict__ widx,
                          const float* __restrict__ p, const float* __restrict__ attn,
                          int S, int B, int V) {
  int i = blockIdx.x * blockDim.x + threadIdx.x;
  if (i >= B * S) return;
  int b = i / S, s = i - b * S;
  int idx = widx[(size_t)s * B + b];
  atomicAdd(out + (size_t)b * V + idx, p[b] * attn[(size_t)b * S + s]);
}

__global__ void log_k(float* __restrict__ out, long total) {
  long i = (long)blockIdx.x * blockDim.x + threadIdx.x;
  if (i < total) out[i] = __logf(out[i]);
}

// ---------------------------------------------------------------------------
extern "C" void kernel_launch(void* const* d_in, const int* in_sizes, int n_in,
                              void* d_out, int out_size, void* d_ws, size_t ws_size,
                              hipStream_t stream) {
  const float* embedded = (const float*)d_in[0];
  const float* hidden   = (const float*)d_in[1];
  const float* enc      = (const float*)d_in[2];
  const int*   widx     = (const int*)d_in[3];
  const float* W_ih     = (const float*)d_in[5];
  const float* W_hh     = (const float*)d_in[6];
  const float* b_ih     = (const float*)d_in[7];
  const float* b_hh     = (const float*)d_in[8];
  const float* W_bil    = (const float*)d_in[9];
  const float* b_bil    = (const float*)d_in[10];
  const float* W_out    = (const float*)d_in[11];
  const float* b_out    = (const float*)d_in[12];
  const float* W_ptr    = (const float*)d_in[13];
  const float* b_ptr    = (const float*)d_in[14];

  const int H = 512, E = 512;
  const int B = in_sizes[0] / E;               // 128
  const int S = in_sizes[2] / (B * H);         // 400
  const int V = in_sizes[11] / (2 * H);        // 50000
  const int SC = (S + 3) / 4;                  // s-chunk size

  // workspace layout (floats): gi, gh, combined, v, energy, stats, partials
  float* ws      = (float*)d_ws;
  float* ws_gi   = ws;
  float* ws_gh   = ws_gi + (size_t)B * 3 * H;
  float* ws_comb = ws_gh + (size_t)B * 3 * H;
  float* ws_v    = ws_comb + (size_t)B * 2 * H;
  float* ws_e    = ws_v + (size_t)B * H;
  float* ws_st   = ws_e + (size_t)B * S;
  float* ws_part = ws_st + (size_t)B * 2;      // B*4*(H+2)

  // d_out layout: out[B,V] | h_new[B,H] | enc_attn[B,S] | prob_ptr[B]
  float* out      = (float*)d_out;
  float* out_h    = out + (size_t)B * V;
  float* out_attn = out_h + (size_t)B * H;
  float* out_p    = out_attn + (size_t)B * S;

  const long totV = (long)B * V;

  // 1-2: GRU gate GEMMs (bf16 WMMA, NT)
  gemm_nt_bf16<<<(3 * H + 63) / 64, 256, 0, stream>>>(embedded, W_ih, b_ih, ws_gi, 3 * H, E);
  gemm_nt_bf16<<<(3 * H + 63) / 64, 256, 0, stream>>>(hidden,   W_hh, b_hh, ws_gh, 3 * H, H);
  // 3: gates -> h_new
  gru_elem<<<(B * H + 255) / 256, 256, 0, stream>>>(ws_gi, ws_gh, hidden, out_h, ws_comb, B, H);
  // 4: bilinear v = h_new @ W_bil
  bil_v<<<B, 256, 0, stream>>>(out_h, W_bil, ws_v, H);
  // 5: fused energy + online softmax + context (single pass over enc)
  attn_fused<<<B * 4, 512, 0, stream>>>(ws_v, enc, b_bil, ws_e, ws_part, S, B, H, SC);
  // 6: merge partials -> context + enc_attn output
  attn_combine<<<B, 512, 0, stream>>>(ws_part, ws_e, out_attn, ws_comb, S, H);
  // 7: pointer probability
  pptr_k<<<B, 256, 0, stream>>>(ws_comb, W_ptr, b_ptr, out_p, 2 * H);
  // 8: big logits GEMM into d_out (bf16 WMMA, inline f32->bf16 of W_out)
  gemm_nt_bf16<<<(V + 63) / 64, 256, 0, stream>>>(ws_comb, W_out, b_out, out, V, 2 * H);
  // 9: row softmax stats over V
  rowstats_k<<<B, 256, 0, stream>>>(out, ws_st, V);
  // 10: (1-p)*softmax in place
  finalize_k<<<(int)((totV + 255) / 256), 256, 0, stream>>>(out, ws_st, out_p, V, totV);
  // 11: pointer scatter-add
  scatter_k<<<(B * S + 255) / 256, 256, 0, stream>>>(out, widx, out_p, out_attn, S, B, V);
  // 12: log
  log_k<<<(int)((totV + 255) / 256), 256, 0, stream>>>(out, totV);
}
